// MUSE_69818988363860
// MI455X (gfx1250) — matrix operand order repaired and checked by reading, not compiled, not run on hardware
//
#include <hip/hip_runtime.h>
#include <hip/hip_bf16.h>

// SR-GNN fused implementation for gfx1250 (MI455X), wave32 + WMMA + async LDS copies.
//
// Kernel 0: f32 -> f16 weight conversion into workspace.
// Kernel 1: per-batch GNN/GRU cell, all GEMMs via v_wmma_f32_16x16x32_f16;
//           embedding gather + adjacency load via global_load_async_to_lds_b128.
// Kernel 2: per-batch attention readout, GEMMs via WMMA, async row gather,
//           fused masked attention reduction.

typedef __attribute__((ext_vector_type(16))) _Float16 v16h;
typedef __attribute__((ext_vector_type(8)))  float    v8f;

#define Bn 1024
#define Nn 50
#define Ln 50
#define Dn 128
#define NP 64   // padded session length (50 -> 64, 4 tiles of 16)

__device__ __forceinline__ float sigmoidf_(float x) {
    return 1.0f / (1.0f + __expf(-x));
}

// Per-lane async global->LDS copy of 16 bytes (GLOBAL_LOAD_ASYNC_TO_LDS_B128,
// tracked by ASYNCcnt; no VGPR data movement).
__device__ __forceinline__ void async_ld_b128(unsigned lds_off, const void* gaddr) {
    asm volatile("global_load_async_to_lds_b128 %0, %1, off"
                 :: "v"(lds_off), "v"(gaddr) : "memory");
}
__device__ __forceinline__ void wait_async0() {
    asm volatile("s_wait_asynccnt 0" ::: "memory");
}
// Generic LDS pointer -> 32-bit LDS byte offset (low 32 bits of the aperture address).
__device__ __forceinline__ unsigned lds_off_of(const void* p) {
    return (unsigned)(unsigned long long)p;
}

// A fragment: 16x32 f16, rows m0..m0+15, K = k0..k0+31. X row-major, leading dim ld.
// Lane L (0..15): M=L, K = 0..15; lane L+16: M=L, K = 16..31.
__device__ __forceinline__ v16h load_a16(const _Float16* X, int ld, int m0, int k0, int lane) {
    const int m  = m0 + (lane & 15);
    const int kh = (lane >> 4) << 4;
    const _Float16* p0 = X + m * ld + k0 + kh;
    const _Float16* p1 = p0 + 8;
    v16h a;
#pragma unroll
    for (int i = 0; i < 8; ++i) { a[i] = p0[i]; a[i + 8] = p1[i]; }
    return a;
}

// B fragment for X @ W^T: B[k][n] = W[n][k]. W row-major [rows][ld].
// Lane n (0..15): col n0+n, K = k0..k0+15; lane n+16: col n0+n, K = k0+16..k0+31.
__device__ __forceinline__ v16h load_bT16(const _Float16* W, int ld, int n0, int k0, int lane) {
    const int n  = n0 + (lane & 15);
    const int kb = (lane >> 4) << 4;
    const _Float16* p = W + n * ld + k0 + kb;
    v16h b;
#pragma unroll
    for (int i = 0; i < 16; ++i) b[i] = p[i];
    return b;
}

__device__ __forceinline__ v8f wmma16(v16h a, v16h b, v8f c) {
    return __builtin_amdgcn_wmma_f32_16x16x32_f16(
        /*neg_a=*/false, a, /*neg_b=*/false, b,
        /*c_mod=*/(short)0, c, /*reuse_a=*/false, /*reuse_b=*/false);
}

// ---------------------------------------------------------------------------
__global__ __launch_bounds__(256) void f32_to_f16_kernel(const float* __restrict__ src,
                                                         _Float16* __restrict__ dst, int n) {
    int i = blockIdx.x * 256 + threadIdx.x;
    if (i < n) dst[i] = (_Float16)src[i];
}

// ---------------------------------------------------------------------------
// One workgroup (8 waves) per batch element. Dynamic LDS = 131072 bytes.
__global__ __launch_bounds__(256) void gnn_cell_kernel(
    const int* __restrict__ items, const float* __restrict__ A,
    const float* __restrict__ emb,
    const float* __restrict__ b_ih, const float* __restrict__ b_hh,
    const float* __restrict__ b_iah, const float* __restrict__ b_oah,
    const float* __restrict__ bi_in, const float* __restrict__ bi_out,
    const _Float16* __restrict__ w_in_h, const _Float16* __restrict__ w_out_h,
    const _Float16* __restrict__ w_ih_h, const _Float16* __restrict__ w_hh_h,
    float* __restrict__ hidden_out)
{
    extern __shared__ char smem[];
    float*    hid32 = (float*)smem;                       // NP*Dn f32   (32 KB)
    _Float16* hid16 = (_Float16*)(smem + NP * Dn * 4);    // NP*Dn f16   (16 KB)
    _Float16* ain   = hid16 + NP * Dn;                    // NP*NP f16   ( 8 KB)
    _Float16* aout  = ain + NP * NP;                      // NP*NP f16   ( 8 KB)
    _Float16* hinT  = aout + NP * NP;                     // Dn*NP f16 [col][row] (16 KB)
    _Float16* houtT = hinT + Dn * NP;                     // Dn*NP f16   (16 KB)
    _Float16* cat16 = houtT + Dn * NP;                    // NP*2Dn f16  (32 KB)
    float*    astg  = (float*)cat16;                      // A staging (50*100 f32 = 20000 B),
                                                          // aliases cat16 (dead in phase 1)

    const int b    = blockIdx.x;
    const int tid  = threadIdx.x;
    const int lane = tid & 31;
    const int wv   = tid >> 5;

    // Phase 1a: async gather of embedding rows straight into LDS (f32).
    // 50 rows * 32 chunks of 16B each = 1600 async b128 ops per block.
    {
        const unsigned hid32_off = lds_off_of(hid32);
        for (int ch = tid; ch < Nn * (Dn / 4); ch += 256) {
            const int r  = ch >> 5;            // row 0..49
            const int c4 = (ch & 31) << 2;     // float col 0,4,...,124
            const int it = items[b * Nn + r];
            const float* g = emb + (size_t)it * Dn + c4;
            async_ld_b128(hid32_off + (unsigned)((r * Dn + c4) * 4), g);
        }
    }
    // Phase 1b: async load of adjacency tile A[b] (50 x 100 f32) into LDS staging.
    {
        const unsigned astg_off = lds_off_of(astg);
        const float* Ab = A + (size_t)b * Nn * (2 * Nn);
        for (int ch = tid; ch < Nn * (2 * Nn) / 4; ch += 256) {   // 1250 chunks
            async_ld_b128(astg_off + (unsigned)(ch * 16), Ab + ch * 4);
        }
    }
    wait_async0();
    __syncthreads();

    // Phase 1c: produce f16 copies + zero padding rows (async engine wrote rows < Nn only).
    for (int idx = tid; idx < NP * Dn; idx += 256) {
        const int r = idx >> 7;
        float v;
        if (r < Nn) v = hid32[idx];
        else { v = 0.0f; hid32[idx] = 0.0f; }
        hid16[idx] = (_Float16)v;
    }
    __syncthreads();   // astg fully readable; also hid16 stable before WMMA
    for (int idx = tid; idx < NP * NP; idx += 256) {
        const int r = idx >> 6, c = idx & (NP - 1);
        float vi = 0.0f, vo = 0.0f;
        if (r < Nn && c < Nn) {
            vi = astg[r * (2 * Nn) + c];
            vo = astg[r * (2 * Nn) + Nn + c];
        }
        ain[idx]  = (_Float16)vi;
        aout[idx] = (_Float16)vo;
    }
    __syncthreads();

    // Phase 2: Hin = hid@w_in^T + bi_in ; Hout = hid@w_out^T + bi_out, stored transposed.
    // 64 tile jobs (2 matrices x 4 Mtiles x 8 Ntiles) over 8 waves -> uniform.
    for (int job = wv; job < 64; job += 8) {
        const bool isOut = job >= 32;
        const int  t  = job & 31;
        const int  m0 = (t >> 3) << 4;
        const int  n0 = (t & 7) << 4;
        const _Float16* W = isOut ? w_out_h : w_in_h;
        v8f c = {};
#pragma unroll
        for (int k0 = 0; k0 < Dn; k0 += 32)
            c = wmma16(load_a16(hid16, Dn, m0, k0, lane),
                       load_bT16(W, Dn, n0, k0, lane), c);
        const int   col  = n0 + (lane & 15);
        const float bias = isOut ? bi_out[col] : bi_in[col];
        _Float16*   T    = isOut ? houtT : hinT;
#pragma unroll
        for (int j = 0; j < 8; ++j) {
            const int row = m0 + j + ((lane >> 4) << 3);
            T[col * NP + row] = (_Float16)(c[j] + bias);
        }
    }
    __syncthreads();

    // Phase 3: cat = [ a_in@Hin + b_iah , a_out@Hout + b_oah ]   (NP x 2Dn)
    // (overwrites the A staging area -- ain/aout f16 copies already made)
    for (int job = wv; job < 64; job += 8) {
        const bool isOut = job >= 32;
        const int  t  = job & 31;
        const int  m0 = (t >> 3) << 4;
        const int  n0 = (t & 7) << 4;
        const _Float16* Aa = isOut ? aout : ain;
        const _Float16* T  = isOut ? houtT : hinT;   // acts as B: T[n][k] = H[k][n]
        v8f c = {};
#pragma unroll
        for (int k0 = 0; k0 < NP; k0 += 32)
            c = wmma16(load_a16(Aa, NP, m0, k0, lane),
                       load_bT16(T, NP, n0, k0, lane), c);
        const int   col  = n0 + (lane & 15);
        const float bias = isOut ? b_oah[col] : b_iah[col];
        const int   cofs = isOut ? Dn : 0;
#pragma unroll
        for (int j = 0; j < 8; ++j) {
            const int row = m0 + j + ((lane >> 4) << 3);
            cat16[row * (2 * Dn) + cofs + col] = (_Float16)(c[j] + bias);
        }
    }
    __syncthreads();

    // Phase 4: GRU gates. Per tile keep 6 f32 accumulators:
    //   gi_{r,i,n} = cat @ w_ih^T   (K=256), gh_{r,i,n} = hid @ w_hh^T (K=128)
    for (int t = wv; t < 32; t += 8) {
        const int m0 = (t >> 3) << 4;
        const int n0 = (t & 7) << 4;
        v8f gr = {}, gi = {}, gn = {};
        v8f hr = {}, hi = {}, hn = {};
#pragma unroll
        for (int k0 = 0; k0 < 2 * Dn; k0 += 32) {
            v16h a = load_a16(cat16, 2 * Dn, m0, k0, lane);
            gr = wmma16(a, load_bT16(w_ih_h, 2 * Dn, n0,          k0, lane), gr);
            gi = wmma16(a, load_bT16(w_ih_h, 2 * Dn, n0 + Dn,     k0, lane), gi);
            gn = wmma16(a, load_bT16(w_ih_h, 2 * Dn, n0 + 2 * Dn, k0, lane), gn);
        }
#pragma unroll
        for (int k0 = 0; k0 < Dn; k0 += 32) {
            v16h a = load_a16(hid16, Dn, m0, k0, lane);
            hr = wmma16(a, load_bT16(w_hh_h, Dn, n0,          k0, lane), hr);
            hi = wmma16(a, load_bT16(w_hh_h, Dn, n0 + Dn,     k0, lane), hi);
            hn = wmma16(a, load_bT16(w_hh_h, Dn, n0 + 2 * Dn, k0, lane), hn);
        }
        const int   col = n0 + (lane & 15);
        const float bir = b_ih[col], bii = b_ih[col + Dn], bin_ = b_ih[col + 2 * Dn];
        const float bhr = b_hh[col], bhi = b_hh[col + Dn], bhn  = b_hh[col + 2 * Dn];
#pragma unroll
        for (int j = 0; j < 8; ++j) {
            const int row = m0 + j + ((lane >> 4) << 3);
            const float rg  = sigmoidf_(gr[j] + bir + hr[j] + bhr);
            const float ig  = sigmoidf_(gi[j] + bii + hi[j] + bhi);
            const float ng  = tanhf(gn[j] + bin_ + rg * (hn[j] + bhn));
            const float old = hid32[row * Dn + col];
            const float h   = ng + ig * (old - ng);
            if (row < Nn)
                hidden_out[((size_t)b * Nn + row) * Dn + col] = h;
        }
    }
}

// ---------------------------------------------------------------------------
// Attention readout. One workgroup per batch. Dynamic LDS = 67328 bytes.
__global__ __launch_bounds__(256) void readout_kernel(
    const int* __restrict__ alias_inputs, const unsigned char* __restrict__ mask,
    const int* __restrict__ lens,
    const float* __restrict__ w1, const float* __restrict__ b1,
    const float* __restrict__ b2, const float* __restrict__ b3,
    const float* __restrict__ wt, const float* __restrict__ bt,
    const _Float16* __restrict__ w2_h, const _Float16* __restrict__ w3_h,
    const float* __restrict__ hidden,
    float* __restrict__ out_seq_hidden, float* __restrict__ out_seq_output)
{
    extern __shared__ char smem[];
    float*    sh32 = (float*)smem;                      // NP*Dn f32 (32 KB)
    _Float16* sh16 = (_Float16*)(smem + NP * Dn * 4);   // NP*Dn f16 (16 KB)
    _Float16* s16  = sh16 + NP * Dn;                    // NP*Dn f16 (16 KB) sigmoid(q1+q2)
    float*    q1L  = (float*)(s16 + NP * Dn);           // Dn
    float*    htL  = q1L + Dn;                          // Dn
    float*    accL = htL + Dn;                          // Dn
    float*    mskL = accL + Dn;                         // NP

    const int b    = blockIdx.x;
    const int tid  = threadIdx.x;
    const int lane = tid & 31;
    const int wv   = tid >> 5;

    // Phase 1: async gather seq_hidden rows (alias indirection) straight into LDS.
    {
        const unsigned sh32_off = lds_off_of(sh32);
        for (int ch = tid; ch < Ln * (Dn / 4); ch += 256) {   // 1600 chunks
            const int l  = ch >> 5;
            const int c4 = (ch & 31) << 2;
            const int r  = alias_inputs[b * Ln + l];
            const float* g = hidden + ((size_t)b * Nn + r) * Dn + c4;
            async_ld_b128(sh32_off + (unsigned)((l * Dn + c4) * 4), g);
        }
    }
    if (tid < NP)  mskL[tid] = (tid < Ln && mask[b * Ln + tid]) ? 1.0f : 0.0f;
    if (tid < Dn)  accL[tid] = 0.0f;
    wait_async0();
    __syncthreads();

    // Phase 1b: f16 copy, zero padding rows, emit seq_hidden output from LDS.
    for (int idx = tid; idx < NP * Dn; idx += 256) {
        const int l = idx >> 7, c = idx & (Dn - 1);
        float v;
        if (l < Ln) {
            v = sh32[idx];
            out_seq_hidden[((size_t)b * Ln + l) * Dn + c] = v;
        } else { v = 0.0f; sh32[idx] = 0.0f; }
        sh16[idx] = (_Float16)v;
    }
    __syncthreads();

    // Phase 2: ht = seq_hidden[lens-1]
    if (tid < Dn) {
        const int last = lens[b] - 1;
        htL[tid] = sh32[last * Dn + tid];
    }
    __syncthreads();

    // Phase 3: q1 = ht @ w1^T + b1 (scalar vector-matrix, one output per thread)
    if (tid < Dn) {
        float s = b1[tid];
        const float* wr = w1 + (size_t)tid * Dn;
        for (int k = 0; k < Dn; ++k) s += htL[k] * wr[k];
        q1L[tid] = s;
    }
    __syncthreads();

    // Phase 4: S = sigmoid(q1 + seq_hidden@w2^T + b2) -> f16, via WMMA (32 tiles / 8 waves)
    for (int t = wv; t < 32; t += 8) {
        const int m0 = (t >> 3) << 4;
        const int n0 = (t & 7) << 4;
        v8f c = {};
#pragma unroll
        for (int k0 = 0; k0 < Dn; k0 += 32)
            c = wmma16(load_a16(sh16, Dn, m0, k0, lane),
                       load_bT16(w2_h, Dn, n0, k0, lane), c);
        const int   col = n0 + (lane & 15);
        const float qq  = q1L[col] + b2[col];
#pragma unroll
        for (int j = 0; j < 8; ++j) {
            const int row = m0 + j + ((lane >> 4) << 3);
            s16[row * Dn + col] = (_Float16)sigmoidf_(c[j] + qq);
        }
    }
    __syncthreads();

    // Phase 5: alp = S@w3^T + b3, fused masked weighted reduction into accL.
    for (int t = wv; t < 32; t += 8) {
        const int m0 = (t >> 3) << 4;
        const int n0 = (t & 7) << 4;
        v8f c = {};
#pragma unroll
        for (int k0 = 0; k0 < Dn; k0 += 32)
            c = wmma16(load_a16(s16, Dn, m0, k0, lane),
                       load_bT16(w3_h, Dn, n0, k0, lane), c);
        const int   col = n0 + (lane & 15);
        const float bb  = b3[col];
        float p = 0.0f;
#pragma unroll
        for (int j = 0; j < 8; ++j) {
            const int row = m0 + j + ((lane >> 4) << 3);
            p += (c[j] + bb) * sh32[row * Dn + col] * mskL[row];
        }
        atomicAdd(&accL[col], p);
    }
    __syncthreads();

    // Phase 6: seq_output = [a, ht] @ wt^T + bt
    if (tid < Dn) {
        float s = bt[tid];
        const float* wr = wt + (size_t)tid * (2 * Dn);
        for (int k = 0; k < Dn; ++k) s += accL[k] * wr[k] + htL[k] * wr[Dn + k];
        out_seq_output[(size_t)b * Dn + tid] = s;
    }
}

// ---------------------------------------------------------------------------
extern "C" void kernel_launch(void* const* d_in, const int* in_sizes, int n_in,
                              void* d_out, int out_size, void* d_ws, size_t ws_size,
                              hipStream_t stream) {
    (void)in_sizes; (void)n_in; (void)out_size; (void)ws_size;

    const int*   items  = (const int*)d_in[0];
    const float* A      = (const float*)d_in[1];
    const int*   alias  = (const int*)d_in[2];
    const unsigned char* mask = (const unsigned char*)d_in[3];
    const int*   lens   = (const int*)d_in[4];
    const float* emb    = (const float*)d_in[5];
    const float* w_ih   = (const float*)d_in[6];
    const float* w_hh   = (const float*)d_in[7];
    const float* b_ih   = (const float*)d_in[8];
    const float* b_hh   = (const float*)d_in[9];
    const float* b_iah  = (const float*)d_in[10];
    const float* b_oah  = (const float*)d_in[11];
    const float* w_in   = (const float*)d_in[12];
    const float* bi_in  = (const float*)d_in[13];
    const float* w_out  = (const float*)d_in[14];
    const float* bi_out = (const float*)d_in[15];
    const float* w1     = (const float*)d_in[16];
    const float* b1     = (const float*)d_in[17];
    const float* w2     = (const float*)d_in[18];
    const float* b2     = (const float*)d_in[19];
    const float* w3     = (const float*)d_in[20];
    const float* b3     = (const float*)d_in[21];
    const float* wt     = (const float*)d_in[22];
    const float* bt     = (const float*)d_in[23];

    // Workspace layout: f16 weights, then new hidden state (f32).
    _Float16* wsh     = (_Float16*)d_ws;
    _Float16* w_in_h  = wsh;
    _Float16* w_out_h = w_in_h  + Dn * Dn;
    _Float16* w_ih_h  = w_out_h + Dn * Dn;
    _Float16* w_hh_h  = w_ih_h  + 3 * Dn * 2 * Dn;
    _Float16* w2_h    = w_hh_h  + 3 * Dn * Dn;
    _Float16* w3_h    = w2_h    + Dn * Dn;
    float*    hidden_new = (float*)(w3_h + Dn * Dn);   // Bn*Nn*Dn f32

    auto cvt = [&](const float* s, _Float16* d, int n) {
        f32_to_f16_kernel<<<(n + 255) / 256, 256, 0, stream>>>(s, d, n);
    };
    cvt(w_in,  w_in_h,  Dn * Dn);
    cvt(w_out, w_out_h, Dn * Dn);
    cvt(w_ih,  w_ih_h,  3 * Dn * 2 * Dn);
    cvt(w_hh,  w_hh_h,  3 * Dn * Dn);
    cvt(w2,    w2_h,    Dn * Dn);
    cvt(w3,    w3_h,    Dn * Dn);

    const size_t lds1 = (size_t)NP * Dn * 4 + (size_t)NP * Dn * 2 + 2u * NP * NP * 2
                      + 2u * Dn * NP * 2 + (size_t)NP * 2 * Dn * 2;   // 131072
    gnn_cell_kernel<<<Bn, 256, lds1, stream>>>(
        items, A, emb, b_ih, b_hh, b_iah, b_oah, bi_in, bi_out,
        w_in_h, w_out_h, w_ih_h, w_hh_h, hidden_new);

    const size_t lds2 = (size_t)NP * Dn * 4 + 2u * NP * Dn * 2 + (3u * Dn + NP) * 4; // 67328
    float* out_seq_hidden = (float*)d_out;
    float* out_seq_output = out_seq_hidden + (size_t)Bn * Ln * Dn;
    readout_kernel<<<Bn, 256, lds2, stream>>>(
        alias, mask, lens, w1, b1, b2, b3, wt, bt, w2_h, w3_h,
        hidden_new, out_seq_hidden, out_seq_output);
}
